// Decoder_3788161155303
// MI455X (gfx1250) — matrix operand order repaired and checked by reading, not compiled, not run on hardware
//
#include <hip/hip_runtime.h>
#include <hip/hip_bf16.h>

#define B_ 2048
#define T_ 64
#define E_ 256
#define D_ 256

typedef float v2f  __attribute__((ext_vector_type(2)));
typedef float v8f  __attribute__((ext_vector_type(8)));
typedef __bf16 v8bf  __attribute__((ext_vector_type(8)));
typedef __bf16 v16bf __attribute__((ext_vector_type(16)));

// ---------------------------------------------------------------------------
// bf16 WMMA GEMM (K fixed = 256): C_bf16[M,N] = A_bf16[M,256] @ W_bf16[N,256]^T (+bias)
// Block: 256 threads = 8 waves; wave -> one 16x16 tile, 2(M)x4(N) per block.
// W tile (64 rows x 256) staged in LDS (32 KB). Used for the big hoisted GEMM.
// ---------------------------------------------------------------------------
__global__ void __launch_bounds__(256)
gemm_wmma_bf16(const __bf16* __restrict__ A, const __bf16* __restrict__ W,
               const float* __restrict__ bias, __bf16* __restrict__ Cdst,
               int M, int N)
{
    extern __shared__ __align__(16) unsigned int smemw[];   // 64 * 128 dwords
    __bf16* smemB = (__bf16*)smemw;

    const int tid  = threadIdx.x;
    const int wave = tid >> 5;
    const int lane = tid & 31;
    const int r    = lane & 15;
    const int half = lane >> 4;

    const int tm     = blockIdx.y * 2 + (wave >> 2);
    const int nBase  = blockIdx.x * 64;
    const int nLocal = (wave & 3) * 16 + r;
    const int nGlob  = nBase + nLocal;
    const int rowA   = tm * 16 + r;

    // stage 64 W rows (bf16, 256 K each) into LDS as dwords, coalesced
    const unsigned int* W32 = (const unsigned int*)W;
    for (int idx = tid; idx < 64 * 128; idx += 256) {
        int nl = idx >> 7;
        int kk = idx & 127;
        smemw[idx] = W32[(size_t)(nBase + nl) * 128 + kk];
    }
    __syncthreads();

    v8f acc = {};
    const __bf16* Ap = A + (size_t)rowA * 256;
#pragma unroll 2
    for (int k = 0; k < 256; k += 32) {
        // A 16x32 frag: lane holds K = {8*half..8*half+7} and {16+8*half..+7}
        v8bf a0 = *(const v8bf*)(Ap + k + half * 8);
        v8bf a1 = *(const v8bf*)(Ap + k + 16 + half * 8);
        v16bf av = __builtin_shufflevector(a0, a1,
                     0,1,2,3,4,5,6,7,8,9,10,11,12,13,14,15);
        // B 32x16 frag (=W^T): lane N=r holds K = {8*half..8*half+7} and {16+8*half..+7}
        v8bf b0 = *(const v8bf*)(smemB + nLocal * 256 + k + half * 8);
        v8bf b1 = *(const v8bf*)(smemB + nLocal * 256 + k + 16 + half * 8);
        v16bf bv = __builtin_shufflevector(b0, b1,
                     0,1,2,3,4,5,6,7,8,9,10,11,12,13,14,15);
        acc = __builtin_amdgcn_wmma_f32_16x16x32_bf16(
                  false, av, false, bv, (short)0, acc, false, false);
    }

    float bval = bias ? bias[nGlob] : 0.f;
#pragma unroll
    for (int i = 0; i < 8; ++i) {
        int row = tm * 16 + half * 8 + i;
        Cdst[(size_t)row * N + nGlob] = (__bf16)(acc[i] + bval);
    }
}

// ---------------------------------------------------------------------------
// f32 WMMA GEMM (full precision; used on the recurrent path):
// C[M,N] = A[M,K] @ W[N,:]^T (+bias0+bias1 + r2X@r2W^T)
// A = [A0 | A1] split along K at kSplit. M%32==0, N%64==0, K%256==0.
// ---------------------------------------------------------------------------
__global__ void __launch_bounds__(256)
gemm_wmma_f32(const float* __restrict__ A0, const float* __restrict__ A1,
              int lda, int kSplit,
              const float* __restrict__ W, int ldw, int wOff,
              const float* __restrict__ bias0, const float* __restrict__ bias1,
              const float* __restrict__ r2W, const float* __restrict__ r2X,
              float* __restrict__ Cdst, int M, int N, int K)
{
    extern __shared__ float smem[];              // 64 * 256 floats = 64 KB
    const int tid  = threadIdx.x;
    const int wave = tid >> 5;
    const int lane = tid & 31;
    const int r    = lane & 15;
    const int half = lane >> 4;

    const int tm     = blockIdx.y * 2 + (wave >> 2);
    const int nBase  = blockIdx.x * 64;
    const int nLocal = (wave & 3) * 16 + r;
    const int nGlob  = nBase + nLocal;
    const int rowA   = tm * 16 + r;

    v8f acc = {};

    for (int k0 = 0; k0 < K; k0 += 256) {
        __syncthreads();
        for (int idx = tid; idx < 64 * 256; idx += 256) {
            int nl = idx >> 8;
            int kk = idx & 255;
            smem[idx] = W[(size_t)(nBase + nl) * ldw + wOff + k0 + kk];
        }
        __syncthreads();

#pragma unroll 4
        for (int k = 0; k < 256; k += 4) {
            const int kg = k0 + k;
            const float* Ap = (kg < kSplit)
                                ? (A0 + (size_t)rowA * lda + kg)
                                : (A1 + (size_t)rowA * lda + (kg - kSplit));
            v2f a = *(const v2f*)(Ap + 2 * half);
            v2f b = *(const v2f*)(&smem[nLocal * 256 + k + 2 * half]);
            acc = __builtin_amdgcn_wmma_f32_16x16x4_f32(
                      false, a, false, b, (short)0, acc, false, false);
        }
    }

    float bval = 0.f;
    if (bias0) bval += bias0[nGlob];
    if (bias1) bval += bias1[nGlob];
    float w0 = 0.f, w1 = 0.f;
    if (r2W) { w0 = r2W[nGlob * 2 + 0]; w1 = r2W[nGlob * 2 + 1]; }

#pragma unroll
    for (int i = 0; i < 8; ++i) {
        int row = tm * 16 + half * 8 + i;
        float v = acc[i] + bval;
        if (r2X) v += r2X[row * 2 + 0] * w0 + r2X[row * 2 + 1] * w1;
        Cdst[(size_t)row * N + nGlob] = v;
    }
}

// ---------------------------------------------------------------------------
// Fused attention step: one block per batch element b (256 threads, e = tid).
// P and enc are bf16 (L2-resident working set).
// ---------------------------------------------------------------------------
__global__ void __launch_bounds__(256)
attn_kernel(const __bf16* __restrict__ P, const __bf16* __restrict__ enc,
            const float* __restrict__ q, const float* __restrict__ w2,
            const float* __restrict__ b2, const float* __restrict__ yhist,
            const float* __restrict__ Wfc, const float* __restrict__ bfc,
            float* __restrict__ ctx, float* __restrict__ ytl)
{
    __shared__ float sPart[64 * 8];
    __shared__ float sScore[64];
    __shared__ float sAlpha[64];
    __shared__ float sRed0[8];
    __shared__ float sRed1[8];

    const int b    = blockIdx.x;
    const int tid  = threadIdx.x;
    const int wave = tid >> 5;
    const int lane = tid & 31;

    const float qv  = q[b * 256 + tid];
    const float w2v = w2[tid];
    const __bf16* Pb = P + (size_t)b * 64 * 256;
    const __bf16* Eb = enc + (size_t)b * 64 * 256;

    for (int t = 0; t < 64; ++t) {
        float v = tanhf((float)Pb[t * 256 + tid] + qv) * w2v;
        for (int off = 16; off; off >>= 1) v += __shfl_xor(v, off, 32);
        if (lane == 0) sPart[t * 8 + wave] = v;
    }
    __syncthreads();
    if (tid < 64) {
        float s = b2[0];
        for (int w = 0; w < 8; ++w) s += sPart[tid * 8 + w];
        sScore[tid] = s;
    }
    __syncthreads();

    float mx = -1e30f;
    for (int t = 0; t < 64; ++t) mx = fmaxf(mx, sScore[t]);
    float sum = 0.f;
    for (int t = 0; t < 64; ++t) sum += __expf(sScore[t] - mx);
    const float inv = 1.0f / sum;
    if (tid < 64) sAlpha[tid] = __expf(sScore[tid] - mx) * inv;
    __syncthreads();

    float cv = 0.f;
    for (int t = 0; t < 64; ++t) cv += sAlpha[t] * (float)Eb[t * 256 + tid];
    ctx[b * 256 + tid] = cv;

    float x0 = Wfc[0 * 320 + tid] * cv;
    float x1 = Wfc[1 * 320 + tid] * cv;
    if (tid < 64) {
        float yh = yhist[b * 64 + tid];
        x0 += Wfc[0 * 320 + 256 + tid] * yh;
        x1 += Wfc[1 * 320 + 256 + tid] * yh;
    }
    for (int off = 16; off; off >>= 1) {
        x0 += __shfl_xor(x0, off, 32);
        x1 += __shfl_xor(x1, off, 32);
    }
    if (lane == 0) { sRed0[wave] = x0; sRed1[wave] = x1; }
    __syncthreads();
    if (tid == 0) {
        float o0 = bfc[0], o1 = bfc[1];
        for (int w = 0; w < 8; ++w) { o0 += sRed0[w]; o1 += sRed1[w]; }
        ytl[b * 2 + 0] = o0;
        ytl[b * 2 + 1] = o1;
    }
}

// ---------------------------------------------------------------------------
__global__ void __launch_bounds__(256)
lstm_cell(const float* __restrict__ gates, float* __restrict__ h,
          float* __restrict__ c)
{
    int idx = blockIdx.x * 256 + threadIdx.x;
    int b = idx >> 8, d = idx & 255;
    const float* g = gates + (size_t)b * 1024;
    float gi = g[d], gf = g[256 + d], gg = g[512 + d], go = g[768 + d];
    float si = 1.f / (1.f + __expf(-gi));
    float sf = 1.f / (1.f + __expf(-gf));
    float so = 1.f / (1.f + __expf(-go));
    float cn = sf * c[idx] + si * tanhf(gg);
    float hn = so * tanhf(cn);
    c[idx] = cn;
    h[idx] = hn;
}

// ---------------------------------------------------------------------------
__global__ void __launch_bounds__(256)
final_kernel(const float* __restrict__ h, const float* __restrict__ ctx,
             const float* __restrict__ Wf, const float* __restrict__ bf,
             float* __restrict__ out)
{
    __shared__ float sRed0[8];
    __shared__ float sRed1[8];
    const int b = blockIdx.x;
    const int tid = threadIdx.x, wave = tid >> 5, lane = tid & 31;
    float hv = h[b * 256 + tid];
    float cv = ctx[b * 256 + tid];
    float x0 = Wf[0 * 512 + tid] * hv + Wf[0 * 512 + 256 + tid] * cv;
    float x1 = Wf[1 * 512 + tid] * hv + Wf[1 * 512 + 256 + tid] * cv;
    for (int off = 16; off; off >>= 1) {
        x0 += __shfl_xor(x0, off, 32);
        x1 += __shfl_xor(x1, off, 32);
    }
    if (lane == 0) { sRed0[wave] = x0; sRed1[wave] = x1; }
    __syncthreads();
    if (tid == 0) {
        float o0 = bf[0], o1 = bf[1];
        for (int w = 0; w < 8; ++w) { o0 += sRed0[w]; o1 += sRed1[w]; }
        out[b * 2 + 0] = o0;
        out[b * 2 + 1] = o1;
    }
}

// elementwise / strided f32 -> bf16 (row length fixed 256)
__global__ void __launch_bounds__(256)
cvt_bf16(const float* __restrict__ src, __bf16* __restrict__ dst,
         int n, int srcStride, int srcOff)
{
    int i = blockIdx.x * 256 + threadIdx.x;
    if (i < n) {
        int row = i >> 8, k = i & 255;
        dst[i] = (__bf16)src[(size_t)row * srcStride + srcOff + k];
    }
}

__global__ void __launch_bounds__(256)
zero_kernel(float* __restrict__ p, int n)
{
    int i = blockIdx.x * 256 + threadIdx.x;
    if (i < n) p[i] = 0.f;
}

// ---------------------------------------------------------------------------
extern "C" void kernel_launch(void* const* d_in, const int* in_sizes, int n_in,
                              void* d_out, int out_size, void* d_ws, size_t ws_size,
                              hipStream_t stream)
{
    const float* enc   = (const float*)d_in[0];   // [B,T,E]
    const float* yhist = (const float*)d_in[1];   // [B,T]
    const float* W1    = (const float*)d_in[2];   // [E,768]
    const float* b1    = (const float*)d_in[3];   // [E]
    const float* w2    = (const float*)d_in[4];   // [E]
    const float* b2    = (const float*)d_in[5];   // [1]
    const float* W_ih  = (const float*)d_in[6];   // [1024,2]
    const float* b_ih  = (const float*)d_in[7];   // [1024]
    const float* W_hh  = (const float*)d_in[8];   // [1024,256]
    const float* b_hh  = (const float*)d_in[9];   // [1024]
    const float* W_fc  = (const float*)d_in[10];  // [2,320]
    const float* b_fc  = (const float*)d_in[11];  // [2]
    const float* W_fin = (const float*)d_in[12];  // [2,512]
    const float* b_fin = (const float*)d_in[13];  // [2]
    float* out = (float*)d_out;

    // workspace carve-up (byte-based; every chunk is >=256B aligned)
    char* wsb = (char*)d_ws;
    __bf16* Pb   = (__bf16*)wsb;                         // [B*T,E] bf16 (67 MB)
    wsb += (size_t)B_ * T_ * E_ * 2;
    __bf16* encb = (__bf16*)wsb;                         // [B*T,E] bf16 (67 MB)
    wsb += (size_t)B_ * T_ * E_ * 2;
    __bf16* W1eb = (__bf16*)wsb;                         // [E,E] bf16
    wsb += (size_t)E_ * E_ * 2;
    float* q     = (float*)wsb;  wsb += (size_t)B_ * E_ * 4;
    float* h     = (float*)wsb;  wsb += (size_t)B_ * D_ * 4;
    float* c     = (float*)wsb;  wsb += (size_t)B_ * D_ * 4;
    float* ctx   = (float*)wsb;  wsb += (size_t)B_ * E_ * 4;
    float* ytl   = (float*)wsb;  wsb += (size_t)B_ * 2 * 4;
    float* gates = (float*)wsb;                           // [B,4D]

    const size_t smemF32  = 64 * 256 * sizeof(float);     // 64 KB
    const size_t smemBF16 = 64 * 256 * sizeof(__bf16);    // 32 KB

    // h = c = 0
    zero_kernel<<<dim3(2 * B_ * D_ / 256), 256, 0, stream>>>(h, 2 * B_ * D_);

    // one-time conversions to bf16
    cvt_bf16<<<dim3(B_ * T_ * E_ / 256), 256, 0, stream>>>(
        enc, encb, B_ * T_ * E_, 256, 0);
    cvt_bf16<<<dim3(E_ * E_ / 256), 256, 0, stream>>>(
        W1, W1eb, E_ * E_, 768, 512);

    // Hoisted precompute on the high-throughput bf16 path:
    // P = enc @ W1[:,512:768]^T + b1   (time-invariant, f32 accumulate)
    gemm_wmma_bf16<<<dim3(E_ / 64, (B_ * T_) / 32), 256, smemBF16, stream>>>(
        encb, W1eb, b1, Pb, B_ * T_, E_);

    for (int step = 0; step < T_; ++step) {
        // q = [h|c] @ W1[:,0:512]^T   (full f32 on the recurrence)
        gemm_wmma_f32<<<dim3(E_ / 64, B_ / 32), 256, smemF32, stream>>>(
            h, c, D_, 256, W1, 768, 0, nullptr, nullptr, nullptr, nullptr,
            q, B_, E_, 512);

        attn_kernel<<<dim3(B_), 256, 0, stream>>>(
            Pb, encb, q, w2, b2, yhist, W_fc, b_fc, ctx, ytl);

        // gates = h @ W_hh^T + (b_ih + b_hh) + y_tilde @ W_ih^T
        gemm_wmma_f32<<<dim3(1024 / 64, B_ / 32), 256, smemF32, stream>>>(
            h, h, D_, 256, W_hh, 256, 0, b_ih, b_hh, W_ih, ytl,
            gates, B_, 1024, 256);

        lstm_cell<<<dim3(B_ * D_ / 256), 256, 0, stream>>>(gates, h, c);
    }

    final_kernel<<<dim3(B_), 256, 0, stream>>>(h, ctx, W_fin, b_fin, out);
}
